// _NeuronGPT2Block_25409026523690
// MI455X (gfx1250) — matrix operand order repaired and checked by reading, not compiled, not run on hardware
//
#include <hip/hip_runtime.h>
#include <hip/hip_bf16.h>
#include <math.h>

// Problem constants (match reference)
#define DI     1024
#define HI     16
#define HDI    64
#define SI     2048
#define BI     2
#define INNERI 4096
#define LOG2S  11
static_assert(SI == (1 << LOG2S), "S must be 2^LOG2S");

typedef __attribute__((ext_vector_type(16))) __bf16   v16bf;
typedef __attribute__((ext_vector_type(8)))  float    v8f;
typedef __attribute__((ext_vector_type(4)))  unsigned u32x4;
typedef __attribute__((ext_vector_type(4)))  int      v4i;

#ifndef __has_builtin
#define __has_builtin(x) 0
#endif
#if __has_builtin(__builtin_amdgcn_global_load_async_to_lds_b128) && \
    __has_builtin(__builtin_amdgcn_s_wait_asynccnt)
#define HAVE_ASYNC 1
using as1_v4i = __attribute__((address_space(1))) v4i;
using as3_v4i = __attribute__((address_space(3))) v4i;
#else
#define HAVE_ASYNC 0
#endif

union FragBF { v16bf v; u32x4 u[2]; };

// A fragment (16x32 MxK, 16-bit): lane<16 -> M=lane, K={kb..kb+7, kb+16..kb+23}, kb=(lane>>4)*8
__device__ __forceinline__ v16bf ld_fragA(const __hip_bfloat16* p, int lane) {
  const int kb = (lane >> 4) * 8;
  FragBF f;
  f.u[0] = *(const u32x4*)(p + kb);
  f.u[1] = *(const u32x4*)(p + kb + 16);
  return f.v;
}
// B fragment (32x16 KxN, 16-bit) from [N][K]-major storage: K={kb..kb+7, kb+16..kb+23}, kb=(lane>>4)*8
__device__ __forceinline__ v16bf ld_fragB(const __hip_bfloat16* p, int lane) {
  const int kb = (lane >> 4) * 8;
  FragBF f;
  f.u[0] = *(const u32x4*)(p + kb);
  f.u[1] = *(const u32x4*)(p + kb + 16);
  return f.v;
}
#define WMMA_BF16(a, b, c) \
  __builtin_amdgcn_wmma_f32_16x16x32_bf16(false, (a), false, (b), (short)0, (c), false, false)

// ---------------------------------------------------------------------------
// mods = scale_shift_table[None] + emb.reshape(B,6,D)
__global__ __launch_bounds__(256) void k_mods(const float* __restrict__ sst,
                                              const float* __restrict__ emb,
                                              float* __restrict__ mods, int n) {
  int i = blockIdx.x * 256 + threadIdx.x;
  if (i < n) mods[i] = sst[i % (6 * DI)] + emb[i];
}

// ---------------------------------------------------------------------------
// f32 [K][N] weight -> bf16 [N][K] (transposed), LDS-tiled 32x32
__global__ __launch_bounds__(256) void k_cvt_w_T(const float* __restrict__ wsrc,
                                                 __hip_bfloat16* __restrict__ wt,
                                                 int K, int N) {
  __shared__ float tile[32][33];
  const int tx = threadIdx.x & 31, ty = threadIdx.x >> 5;  // 32 x 8
  const int n0 = blockIdx.x * 32, k0 = blockIdx.y * 32;
#pragma unroll
  for (int i = 0; i < 32; i += 8)
    tile[ty + i][tx] = wsrc[(size_t)(k0 + ty + i) * N + n0 + tx];
  __syncthreads();
#pragma unroll
  for (int i = 0; i < 32; i += 8)
    wt[(size_t)(n0 + ty + i) * K + k0 + tx] = __float2bfloat16(tile[tx][ty + i]);
}

// ---------------------------------------------------------------------------
// out_bf16[row] = (LN(x[row])*w+b) * (1+sc) + sh   (one block per row, D=1024)
__global__ __launch_bounds__(256) void k_ln_mod(const float* __restrict__ x,
                                                const float* __restrict__ lnw,
                                                const float* __restrict__ lnb,
                                                const float* __restrict__ mods,
                                                int whichSh,
                                                __hip_bfloat16* __restrict__ out) {
  __shared__ float red[16];
  const int row = blockIdx.x;
  const int batch = row >> LOG2S;
  const float* xr = x + (size_t)row * DI;
  float s = 0.f, s2 = 0.f;
  for (int i = threadIdx.x; i < DI; i += 256) {
    float v = xr[i];
    s += v; s2 += v * v;
  }
#pragma unroll
  for (int off = 16; off > 0; off >>= 1) {
    s  += __shfl_down(s, off);
    s2 += __shfl_down(s2, off);
  }
  const int wave = threadIdx.x >> 5, lane = threadIdx.x & 31;
  if (lane == 0) { red[wave] = s; red[8 + wave] = s2; }
  __syncthreads();
  float S1 = 0.f, S2 = 0.f;
#pragma unroll
  for (int i = 0; i < 8; i++) { S1 += red[i]; S2 += red[8 + i]; }
  const float mu = S1 * (1.0f / DI);
  const float var = S2 * (1.0f / DI) - mu * mu;
  const float rs = rsqrtf(var + 1e-5f);
  const float* sh = mods + (size_t)batch * 6 * DI + (size_t)whichSh * DI;
  const float* sc = sh + DI;
  for (int i = threadIdx.x; i < DI; i += 256) {
    float xh = (xr[i] - mu) * rs * lnw[i] + lnb[i];
    out[(size_t)row * DI + i] = __float2bfloat16(xh * (1.f + sc[i]) + sh[i]);
  }
}

// ---------------------------------------------------------------------------
// bf16 GEMM: C[M,N] = A[M,K] @ Wt[N,K]^T + bias.  Block tile 128x128, BK=64,
// 8 waves (4 along M x 2 along N), each wave 32x64 = 8 wmma accumulators,
// 16 wmma issues between barrier pairs.  Staging:
//   HAVE_ASYNC: global_load_async_to_lds_b128 double-buffered (ASYNCcnt pipeline)
//   fallback  : register-staged, next tile loaded while current computes
// EPI 0: f32 out;  EPI 1: gelu -> bf16 out;  EPI 2: f32 out = t*gate + res
#define LDT 72  // LDS row stride in halfs (144B, 16B-aligned, conflict-skewed)
#define BKI 64
template <int EPI>
__global__ __launch_bounds__(256, 2)
void k_gemm(const __hip_bfloat16* __restrict__ A,
            const __hip_bfloat16* __restrict__ Bt,
            const float* __restrict__ bias,
            float* __restrict__ outF,
            __hip_bfloat16* __restrict__ outH,
            const float* __restrict__ gate,
            const float* __restrict__ res,
            int N, int K) {
#if HAVE_ASYNC
  __shared__ __hip_bfloat16 As[2][128 * LDT];
  __shared__ __hip_bfloat16 Bs[2][128 * LDT];
#else
  __shared__ __hip_bfloat16 As[1][128 * LDT];
  __shared__ __hip_bfloat16 Bs[1][128 * LDT];
#endif
  const int tid = threadIdx.x;
  const int wave = tid >> 5, lane = tid & 31;
  const int wm = wave & 3, wn = wave >> 2;
  const int rowBase = blockIdx.y * 128;
  const int colBase = blockIdx.x * 128;
  const int hl = lane & 15;

  v8f acc[2][4] = {};

  const int ldRow = tid >> 1;          // 0..127 (2 threads per row)
  const int ldSeg = (tid & 1) * 32;    // halfs: each thread stages 32 halfs = 4x b128
  const __hip_bfloat16* gA = A + (size_t)(rowBase + ldRow) * K + ldSeg;
  const __hip_bfloat16* gB = Bt + (size_t)(colBase + ldRow) * K + ldSeg;
  __hip_bfloat16* sAp = &As[0][ldRow * LDT + ldSeg];
  __hip_bfloat16* sBp = &Bs[0][ldRow * LDT + ldSeg];

  auto compute = [&](const __hip_bfloat16* Ab, const __hip_bfloat16* Bb) {
#pragma unroll
    for (int ks = 0; ks < 2; ks++) {
      v16bf af0 = ld_fragA(Ab + (wm * 32 + hl) * LDT + ks * 32, lane);
      v16bf af1 = ld_fragA(Ab + (wm * 32 + 16 + hl) * LDT + ks * 32, lane);
#pragma unroll
      for (int nt = 0; nt < 4; nt++) {
        v16bf bfr = ld_fragB(Bb + (wn * 64 + nt * 16 + hl) * LDT + ks * 32, lane);
        acc[0][nt] = WMMA_BF16(af0, bfr, acc[0][nt]);
        acc[1][nt] = WMMA_BF16(af1, bfr, acc[1][nt]);
      }
    }
  };

#if HAVE_ASYNC
  auto issue = [&](int buf, int kk) {
    as3_v4i* la = (as3_v4i*)(void*)(sAp + buf * 128 * LDT);
    as3_v4i* lb = (as3_v4i*)(void*)(sBp + buf * 128 * LDT);
    as1_v4i* ga = (as1_v4i*)(void*)(gA + kk);
    as1_v4i* gb = (as1_v4i*)(void*)(gB + kk);
    __builtin_amdgcn_global_load_async_to_lds_b128(ga, la, 0, 0);
    __builtin_amdgcn_global_load_async_to_lds_b128(ga, la, 16, 0);
    __builtin_amdgcn_global_load_async_to_lds_b128(ga, la, 32, 0);
    __builtin_amdgcn_global_load_async_to_lds_b128(ga, la, 48, 0);
    __builtin_amdgcn_global_load_async_to_lds_b128(gb, lb, 0, 0);
    __builtin_amdgcn_global_load_async_to_lds_b128(gb, lb, 16, 0);
    __builtin_amdgcn_global_load_async_to_lds_b128(gb, lb, 32, 0);
    __builtin_amdgcn_global_load_async_to_lds_b128(gb, lb, 48, 0);
  };
  issue(0, 0);
  int buf = 0;
  for (int kk = 0; kk < K; kk += BKI) {
    __syncthreads();  // everyone done reading buf^1 from two iterations ago
    if (kk + BKI < K) {
      issue(buf ^ 1, kk + BKI);
      __builtin_amdgcn_s_wait_asynccnt(8);  // in-order: current tile's 8 loads landed
    } else {
      __builtin_amdgcn_s_wait_asynccnt(0);
    }
    __syncthreads();  // all waves' current-tile data visible in LDS
    compute(&As[buf][0], &Bs[buf][0]);
    buf ^= 1;
  }
#else
  u32x4 ra[4], rb[4];
  {
    const u32x4* pa = (const u32x4*)gA;
    const u32x4* pb = (const u32x4*)gB;
    ra[0] = pa[0]; ra[1] = pa[1]; ra[2] = pa[2]; ra[3] = pa[3];
    rb[0] = pb[0]; rb[1] = pb[1]; rb[2] = pb[2]; rb[3] = pb[3];
  }
  for (int kk = 0; kk < K; kk += BKI) {
    __syncthreads();  // previous compute done reading LDS
    {
      u32x4* da = (u32x4*)sAp;
      u32x4* db = (u32x4*)sBp;
      da[0] = ra[0]; da[1] = ra[1]; da[2] = ra[2]; da[3] = ra[3];
      db[0] = rb[0]; db[1] = rb[1]; db[2] = rb[2]; db[3] = rb[3];
    }
    if (kk + BKI < K) {  // stage next tile; latency hidden behind the 16 wmma below
      const u32x4* pa = (const u32x4*)(gA + kk + BKI);
      const u32x4* pb = (const u32x4*)(gB + kk + BKI);
      ra[0] = pa[0]; ra[1] = pa[1]; ra[2] = pa[2]; ra[3] = pa[3];
      rb[0] = pb[0]; rb[1] = pb[1]; rb[2] = pb[2]; rb[3] = pb[3];
      __builtin_prefetch((const void*)(pa + 8), 0, 1);
      __builtin_prefetch((const void*)(pb + 8), 0, 1);
    }
    __syncthreads();
    compute(&As[0][0], &Bs[0][0]);
  }
#endif

  const int grp8 = (lane >> 4) * 8;
  const int cl = lane & 15;
#pragma unroll
  for (int mt = 0; mt < 2; mt++)
#pragma unroll
    for (int nt = 0; nt < 4; nt++) {
      const int col = colBase + wn * 64 + nt * 16 + cl;
      const float bv = bias[col];
#pragma unroll
      for (int r = 0; r < 8; r++) {
        const int row = rowBase + wm * 32 + mt * 16 + grp8 + r;
        float t = acc[mt][nt][r] + bv;
        if constexpr (EPI == 0) {
          outF[(size_t)row * N + col] = t;
        } else if constexpr (EPI == 1) {
          float g = 0.5f * t * (1.f + tanhf(0.7978845608028654f * (t + 0.044715f * t * t * t)));
          outH[(size_t)row * N + col] = __float2bfloat16(g);
        } else {
          float gv = gate[(row >> LOG2S) * (6 * DI) + col];
          outF[(size_t)row * N + col] = t * gv + res[(size_t)row * N + col];
        }
      }
    }
}

// ---------------------------------------------------------------------------
// RoPE + repack: qkv f32 [B*S, 3D] -> q,k bf16 [B,H,S,HD], v bf16 transposed [B,H,HD,S]
__global__ __launch_bounds__(256) void k_rope(const float* __restrict__ qkv,
                                              const float* __restrict__ rcos,
                                              const float* __restrict__ rsin,
                                              __hip_bfloat16* __restrict__ q,
                                              __hip_bfloat16* __restrict__ k,
                                              __hip_bfloat16* __restrict__ vt) {
  const int idx = blockIdx.x * 256 + threadIdx.x;  // (b*S+s)*H + h
  const int h = idx % HI;
  const int bs = idx / HI;
  const int s = bs & (SI - 1);
  const int b = bs >> LOG2S;
  const float* base = qkv + (size_t)bs * (3 * DI);
  const float* qp = base + h * HDI;
  const float* kp = base + DI + h * HDI;
  const float* vp = base + 2 * DI + h * HDI;
  const float* cs = rcos + (size_t)s * (HDI / 2);
  const float* sn = rsin + (size_t)s * (HDI / 2);
  const size_t headOff = ((size_t)(b * HI + h) * SI + s) * HDI;
  __hip_bfloat16* qo = q + headOff;
  __hip_bfloat16* ko = k + headOff;
#pragma unroll
  for (int i = 0; i < HDI / 2; i++) {
    const float c = cs[i], sv = sn[i];
    float e = qp[2 * i], o = qp[2 * i + 1];
    qo[2 * i]     = __float2bfloat16(e * c - o * sv);
    qo[2 * i + 1] = __float2bfloat16(e * sv + o * c);
    e = kp[2 * i]; o = kp[2 * i + 1];
    ko[2 * i]     = __float2bfloat16(e * c - o * sv);
    ko[2 * i + 1] = __float2bfloat16(e * sv + o * c);
  }
  const size_t vtBase = (size_t)(b * HI + h) * HDI * SI;
#pragma unroll
  for (int i = 0; i < HDI; i++)
    vt[vtBase + (size_t)i * SI + s] = __float2bfloat16(vp[i]);
}

// ---------------------------------------------------------------------------
// Flash attention: 64-query tile per block (4 waves x 16 rows), 64-key blocks.
// QK^T and P@V via v_wmma_f32_16x16x32_bf16; streaming softmax in f32.
#define LDK 72
__global__ __launch_bounds__(128, 2)
void k_flash(const __hip_bfloat16* __restrict__ Q,
             const __hip_bfloat16* __restrict__ Kd,
             const __hip_bfloat16* __restrict__ Vt,
             __hip_bfloat16* __restrict__ out) {
  __shared__ __hip_bfloat16 Ks[64 * LDK];   // [key][hd]
  __shared__ __hip_bfloat16 Vs[64 * LDK];   // [hd][key]
  __shared__ __hip_bfloat16 Ps[64 * LDK];   // [query][key]
  const int qt = blockIdx.x, h = blockIdx.y, b = blockIdx.z;
  const int bh = b * HI + h;
  const int tid = threadIdx.x;
  const int wave = tid >> 5, lane = tid & 31;
  const int hl = lane & 15, grp = lane >> 4;
  const int grp8 = grp * 8, cl = hl;

  // Q fragments live in registers (16 rows per wave, K = HD = 64 -> 2 k-steps)
  const __hip_bfloat16* qbase = Q + ((size_t)bh * SI + (size_t)qt * 64) * HDI;
  v16bf aq[2];
#pragma unroll
  for (int ks = 0; ks < 2; ks++)
    aq[ks] = ld_fragA(qbase + (size_t)(wave * 16 + hl) * HDI + ks * 32, lane);

  v8f o[4] = {};
  float mstat[8], lstat[8];
#pragma unroll
  for (int r = 0; r < 8; r++) { mstat[r] = -3.0e38f; lstat[r] = 0.f; }

  const int ldr = tid >> 1, ldseg = (tid & 1) * 32;  // 64 rows x 64 halfs coop load
  const int nblk = qt + 1;                           // causal: only key blocks <= qt
  for (int kbi = 0; kbi < nblk; kbi++) {
    __syncthreads();
    {
      const u32x4* gk = (const u32x4*)(Kd + ((size_t)bh * SI + (size_t)kbi * 64 + ldr) * HDI) + (ldseg >> 3);
      u32x4* sk = (u32x4*)(Ks + ldr * LDK + ldseg);
      sk[0] = gk[0]; sk[1] = gk[1]; sk[2] = gk[2]; sk[3] = gk[3];
      const u32x4* gv = (const u32x4*)(Vt + ((size_t)bh * HDI + ldr) * SI + (size_t)kbi * 64) + (ldseg >> 3);
      u32x4* sv = (u32x4*)(Vs + ldr * LDK + ldseg);
      sv[0] = gv[0]; sv[1] = gv[1]; sv[2] = gv[2]; sv[3] = gv[3];
    }
    __syncthreads();

    // scores S = Q @ K^T   (16x64 per wave)
    v8f sacc[4] = {};
#pragma unroll
    for (int nt = 0; nt < 4; nt++)
#pragma unroll
      for (int ks = 0; ks < 2; ks++) {
        v16bf bk = ld_fragB(Ks + (nt * 16 + hl) * LDK + ks * 32, lane);
        sacc[nt] = WMMA_BF16(aq[ks], bk, sacc[nt]);
      }

    // streaming softmax (rows r+8*grp live in the 16-lane group)
#pragma unroll
    for (int r = 0; r < 8; r++) {
      const int qrow = qt * 64 + wave * 16 + grp8 + r;
      float mx = mstat[r];
#pragma unroll
      for (int nt = 0; nt < 4; nt++) {
        const int key = kbi * 64 + nt * 16 + cl;
        float v = sacc[nt][r] * 0.125f + (key > qrow ? -1.0e9f : 0.f);
        sacc[nt][r] = v;
        mx = fmaxf(mx, v);
      }
#pragma unroll
      for (int off = 1; off < 16; off <<= 1) mx = fmaxf(mx, __shfl_xor(mx, off, 16));
      const float alpha = __expf(mstat[r] - mx);
      float ssum = 0.f;
#pragma unroll
      for (int nt = 0; nt < 4; nt++) {
        float p = __expf(sacc[nt][r] - mx);
        sacc[nt][r] = p;
        ssum += p;
      }
#pragma unroll
      for (int off = 1; off < 16; off <<= 1) ssum += __shfl_xor(ssum, off, 16);
      lstat[r] = lstat[r] * alpha + ssum;
      mstat[r] = mx;
#pragma unroll
      for (int nt = 0; nt < 4; nt++) o[nt][r] *= alpha;
    }

    // stage P as bf16 A-operand in LDS
#pragma unroll
    for (int nt = 0; nt < 4; nt++)
#pragma unroll
      for (int r = 0; r < 8; r++)
        Ps[(wave * 16 + grp8 + r) * LDK + nt * 16 + cl] = __float2bfloat16(sacc[nt][r]);
    __syncthreads();

    // O += P @ V  (K dim = 64 keys -> 2 k-steps)
#pragma unroll
    for (int ks = 0; ks < 2; ks++) {
      v16bf ap = ld_fragA(Ps + (wave * 16 + hl) * LDK + ks * 32, lane);
#pragma unroll
      for (int nt = 0; nt < 4; nt++) {
        v16bf bv = ld_fragB(Vs + (nt * 16 + hl) * LDK + ks * 32, lane);
        o[nt] = WMMA_BF16(ap, bv, o[nt]);
      }
    }
  }

  // normalize and scatter to [B,S,D] at columns h*HD + ...
#pragma unroll
  for (int r = 0; r < 8; r++) {
    const float inv = 1.f / lstat[r];
    const int srow = qt * 64 + wave * 16 + grp8 + r;
#pragma unroll
    for (int nt = 0; nt < 4; nt++)
      out[((size_t)b * SI + srow) * DI + h * HDI + nt * 16 + cl] =
          __float2bfloat16(o[nt][r] * inv);
  }
}

// ---------------------------------------------------------------------------
extern "C" void kernel_launch(void* const* d_in, const int* in_sizes, int n_in,
                              void* d_out, int out_size, void* d_ws, size_t ws_size,
                              hipStream_t stream) {
  (void)in_sizes; (void)n_in; (void)out_size; (void)ws_size;
  const float* hidden = (const float*)d_in[0];
  // d_in[1] attention_mask: causal mask is recomputed analytically in k_flash
  const float* emb  = (const float*)d_in[2];
  const float* rcos = (const float*)d_in[3];
  const float* rsin = (const float*)d_in[4];
  const float* ln1w = (const float*)d_in[5];
  const float* ln1b = (const float*)d_in[6];
  const float* ln2w = (const float*)d_in[7];
  const float* ln2b = (const float*)d_in[8];
  const float* attnW = (const float*)d_in[9];
  const float* attnB = (const float*)d_in[10];
  const float* projW = (const float*)d_in[11];
  const float* projB = (const float*)d_in[12];
  const float* fcW  = (const float*)d_in[13];
  const float* fcB  = (const float*)d_in[14];
  const float* p2W  = (const float*)d_in[15];
  const float* p2B  = (const float*)d_in[16];
  const float* sst  = (const float*)d_in[17];
  float* outp = (float*)d_out;

  const int M = BI * SI;  // 4096 rows
  char* wsb = (char*)d_ws;
  size_t off = 0;
  auto take = [&](size_t bytes) -> void* {
    void* p = wsb + off;
    off += (bytes + 255) & ~(size_t)255;
    return p;
  };
  __hip_bfloat16* wAttn = (__hip_bfloat16*)take((size_t)DI * 3 * DI * 2);
  __hip_bfloat16* wProj = (__hip_bfloat16*)take((size_t)DI * DI * 2);
  __hip_bfloat16* wFc   = (__hip_bfloat16*)take((size_t)DI * INNERI * 2);
  __hip_bfloat16* wP2   = (__hip_bfloat16*)take((size_t)INNERI * DI * 2);
  float* mods           = (float*)take((size_t)BI * 6 * DI * 4);
  __hip_bfloat16* xhat  = (__hip_bfloat16*)take((size_t)M * DI * 2);
  float* qkv            = (float*)take((size_t)M * 3 * DI * 4);
  __hip_bfloat16* qh    = (__hip_bfloat16*)take((size_t)M * DI * 2);
  __hip_bfloat16* kh    = (__hip_bfloat16*)take((size_t)M * DI * 2);
  __hip_bfloat16* vth   = (__hip_bfloat16*)take((size_t)M * DI * 2);
  __hip_bfloat16* aout  = (__hip_bfloat16*)take((size_t)M * DI * 2);
  float* hidden2        = (float*)qh;            // reuse q/k region (free after flash)
  __hip_bfloat16* gelu  = (__hip_bfloat16*)qkv;  // reuse qkv region (free after rope)

  // 1) modulation vectors
  k_mods<<<dim3((BI * 6 * DI) / 256), 256, 0, stream>>>(sst, emb, mods, BI * 6 * DI);
  // 2) weight convert + transpose to bf16 [N][K]
  k_cvt_w_T<<<dim3(3 * DI / 32, DI / 32), 256, 0, stream>>>(attnW, wAttn, DI, 3 * DI);
  k_cvt_w_T<<<dim3(DI / 32, DI / 32), 256, 0, stream>>>(projW, wProj, DI, DI);
  k_cvt_w_T<<<dim3(INNERI / 32, DI / 32), 256, 0, stream>>>(fcW, wFc, DI, INNERI);
  k_cvt_w_T<<<dim3(DI / 32, INNERI / 32), 256, 0, stream>>>(p2W, wP2, INNERI, DI);
  // 3) LN1 + msa modulation -> bf16
  k_ln_mod<<<dim3(M), 256, 0, stream>>>(hidden, ln1w, ln1b, mods, 0, xhat);
  // 4) QKV GEMM (f32 out)
  k_gemm<0><<<dim3(3 * DI / 128, M / 128), 256, 0, stream>>>(
      xhat, wAttn, attnB, qkv, nullptr, nullptr, nullptr, 3 * DI, DI);
  // 5) RoPE + head repack (q,k [B,H,S,HD]; v^T [B,H,HD,S]) in bf16
  k_rope<<<dim3((M * HI) / 256), 256, 0, stream>>>(qkv, rcos, rsin, qh, kh, vth);
  // 6) flash attention -> bf16 [B,S,D]
  k_flash<<<dim3(SI / 64, HI, BI), 128, 0, stream>>>(qh, kh, vth, aout);
  // 7) attn proj GEMM, epilogue: *g_msa + residual(hidden_states) -> hidden2 f32
  k_gemm<2><<<dim3(DI / 128, M / 128), 256, 0, stream>>>(
      aout, wProj, projB, hidden2, nullptr, mods + 2 * DI, hidden, DI, DI);
  // 8) LN2 + mlp modulation -> bf16
  k_ln_mod<<<dim3(M), 256, 0, stream>>>(hidden2, ln2w, ln2b, mods, 3, xhat);
  // 9) fc GEMM + GELU -> bf16
  k_gemm<1><<<dim3(INNERI / 128, M / 128), 256, 0, stream>>>(
      xhat, wFc, fcB, nullptr, gelu, nullptr, nullptr, INNERI, DI);
  // 10) mlp proj GEMM, epilogue: *g_mlp + hidden2 -> final f32 output
  k_gemm<2><<<dim3(DI / 128, M / 128), 256, 0, stream>>>(
      gelu, wP2, p2B, outp, nullptr, mods + 5 * DI, hidden2, DI, INNERI);
}
